// Attention_76124000354435
// MI455X (gfx1250) — matrix operand order repaired and checked by reading, not compiled, not run on hardware
//
#include <hip/hip_runtime.h>
#include <hip/hip_bf16.h>
#include <math.h>

typedef __attribute__((ext_vector_type(16))) _Float16 v16h;
typedef __attribute__((ext_vector_type(8)))  _Float16 v8h;
typedef __attribute__((ext_vector_type(8)))  float    v8f;

#define T_SEQ   1024
#define DMODEL  1024
#define NHEAD   16
#define NBR     4
#define DH      64
#define HTOT    64
#define HQDIM   4096   // H_TOT * DH

// ---------------------------------------------------------------------------
// WMMA fragment helpers (layouts per CDNA5 ISA 7.12.2, wave32)
// ---------------------------------------------------------------------------

// 16-bit A matrix 16x32 (MxK), row-major source. Lane data is one contiguous
// 16-half run: K = [16*half, 16*half+16)  -> 2xb128.
__device__ __forceinline__ v16h load_a_frag(const _Float16* __restrict__ A,
                                            int lda, int row0, int k0) {
  int lane = threadIdx.x & 31;
  int m = lane & 15, half = lane >> 4;
  const _Float16* p = A + (size_t)(row0 + m) * lda + k0 + 16 * half;
  v8h lo = *(const v8h*)p;
  v8h hi = *(const v8h*)(p + 8);
  v16h a;
#pragma unroll
  for (int i = 0; i < 8; ++i) { a[i] = lo[i]; a[i + 8] = hi[i]; }
  return a;
}

// B fragment from a TRANSPOSED operand Bt (N x K row-major), so that
// b[j] = B[k0+kh*16+j][n] = Bt[n][k0+kh*16+j] is 32 contiguous bytes per lane.
__device__ __forceinline__ v16h load_bt_frag(const _Float16* __restrict__ Bt,
                                             int ldbt, int k0, int col0) {
  int lane = threadIdx.x & 31;
  int nn = lane & 15, kh = lane >> 4;
  return *(const v16h*)(Bt + (size_t)(col0 + nn) * ldbt + k0 + kh * 16);
}

__device__ __forceinline__ v8f wmma_f16(v16h a, v16h b, v8f c) {
  return __builtin_amdgcn_wmma_f32_16x16x32_f16(false, a, false, b,
                                                (short)0, c, false, false);
}

// ---------------------------------------------------------------------------
// f32 -> f16 convert (plain, for X)
// ---------------------------------------------------------------------------
__global__ void cvt_f16(const float* __restrict__ src, _Float16* __restrict__ dst, int n) {
  int i = blockIdx.x * blockDim.x + threadIdx.x;
  if (i < n) dst[i] = (_Float16)src[i];
}

// ---------------------------------------------------------------------------
// f32 -> f16 convert + transpose: src (rows x cols) f32  ->  dst (cols x rows) f16
// LDS-tiled 32x32, block (32,8).  grid = (cols/32, rows/32)
// ---------------------------------------------------------------------------
__global__ void cvt_f16_t(const float* __restrict__ src, int rows, int cols,
                          _Float16* __restrict__ dst) {
  __shared__ _Float16 tile[32][33];
  int c0 = blockIdx.x * 32, r0 = blockIdx.y * 32;
  int tx = threadIdx.x, ty = threadIdx.y;
#pragma unroll
  for (int i = 0; i < 32; i += 8)
    tile[ty + i][tx] = (_Float16)src[(size_t)(r0 + ty + i) * cols + c0 + tx];
  __syncthreads();
#pragma unroll
  for (int i = 0; i < 32; i += 8)
    dst[(size_t)(c0 + ty + i) * rows + r0 + tx] = tile[tx][ty + i];
}

// ---------------------------------------------------------------------------
// WMMA GEMM: C(MxN) = scale * (A f16 MxK @ Bt^T) + bias,  Bt is (N x K) f16.
// One wave per block; each wave computes a 16x64 strip (A frag reused x4).
// grid = (N/64, M/16). kmask folds the 4-branch mean in the last GEMM
// (chunks are 32-aligned so masking the chunk base is exact).
// transC: write Ch[(col)*ldc + m] instead of Ch[m*ldc + col] (per-head V^T).
// ---------------------------------------------------------------------------
__global__ void gemm_wmma(const _Float16* __restrict__ A,
                          const _Float16* __restrict__ Bt,
                          const float* __restrict__ bias,
                          float* __restrict__ Cf,
                          _Float16* __restrict__ Ch,
                          int K, int lda, int ldbt, int ldc,
                          int kmask, float scale, int transC) {
  int n0 = blockIdx.x * 64;
  int m0 = blockIdx.y * 16;
  v8f zero = {};
  v8f acc[4] = {zero, zero, zero, zero};
  for (int k0 = 0; k0 < K; k0 += 32) {
    int k0m = k0 & kmask;
    v16h a = load_a_frag(A, lda, m0, k0);
#pragma unroll
    for (int nt = 0; nt < 4; ++nt) {
      v16h b = load_bt_frag(Bt, ldbt, k0m, n0 + nt * 16);
      acc[nt] = wmma_f16(a, b, acc[nt]);
    }
  }
  int lane = threadIdx.x & 31;
  int n = lane & 15, half = lane >> 4;
#pragma unroll
  for (int nt = 0; nt < 4; ++nt) {
    int col = n0 + nt * 16 + n;
    float bb = bias ? bias[col] : 0.f;
#pragma unroll
    for (int r = 0; r < 8; ++r) {
      int m = m0 + r + half * 8;
      float v = acc[nt][r] * scale + bb;
      if (Ch) {
        if (transC) Ch[(size_t)col * ldc + m] = (_Float16)v;
        else        Ch[(size_t)m * ldc + col] = (_Float16)v;
      } else {
        Cf[(size_t)m * ldc + col] = v;
      }
    }
  }
}

// ---------------------------------------------------------------------------
// RoPE (NeoX split-half form of the reference) + per-key denom.
// grid = (T, nheads), block = 32. Thread j handles pair (2j, 2j+1) producing
// out[j] (cos branch) and out[j+32] (sin branch).
// dst layout: (head, T, 64) f16 -> attention fragments are contiguous.
// ---------------------------------------------------------------------------
__global__ void rope_kernel(const float* __restrict__ src, int src_ld,
                            _Float16* __restrict__ dst,
                            float* __restrict__ denomOut) {
  int t = blockIdx.x;
  int h = blockIdx.y;
  int j = threadIdx.x & 31;
  const float2* p = (const float2*)(src + (size_t)t * src_ld + h * DH);
  float2 x = p[j];
  // inv_freq = 10000^(-2j/64)
  float inv = __expf(-(float)(2 * j) * (9.210340371976184f / 64.f)); // ln(1e4)
  float ang = (float)t * inv;
  float c = cosf(ang), s = sinf(ang);
  float o1 = x.x * c - x.y * s;   // first half
  float o2 = x.x * s + x.y * c;   // second half
  _Float16* q = dst + ((size_t)h * T_SEQ + t) * DH;
  q[j]      = (_Float16)o1;
  q[j + 32] = (_Float16)o2;
  if (denomOut) {
    float ks = o1 * o1 + o2 * o2;
    ks += __shfl_xor(ks, 16);
    ks += __shfl_xor(ks, 8);
    ks += __shfl_xor(ks, 4);
    ks += __shfl_xor(ks, 2);
    ks += __shfl_xor(ks, 1);
    if (j == 0) denomOut[(size_t)h * T_SEQ + t] = sqrtf(fmaxf(ks, 1e-6f));
  }
}

// ---------------------------------------------------------------------------
// Streaming attention. One wave per (query-tile qt, head h).
// Scores via 2 WMMAs per 16-key tile (K-dim = DH = 64), elementwise
// softplus/SiLU weight (no running max: normalization is a trailing rescale),
// w staged via LDS into an A-fragment, then 4 WMMAs accumulate w @ V using
// the per-head transposed V (contiguous b128 B-fragments).
// ---------------------------------------------------------------------------
__global__ void attn_kernel(const _Float16* __restrict__ Qh,    // (64, T, 64)
                            const _Float16* __restrict__ Kh,    // (16, T, 64)
                            const _Float16* __restrict__ Vt,    // (4096, T)
                            const float* __restrict__ denom,    // (16, T)
                            const float* __restrict__ sink_scalars, // (64)
                            const float* __restrict__ v_nulls,  // (64, 64)
                            _Float16* __restrict__ ctxh) {      // (T, 4096)
  const float ATTN_SCALE = 0.125f;                 // 64^-0.5
  const float GATE = 1.8137993642342178f;          // pi / sqrt(3)
  int qt = blockIdx.x;   // 0..63 query tile (16 rows)
  int h  = blockIdx.y;   // 0..63 head
  int lane = threadIdx.x & 31;
  int n = lane & 15, half = lane >> 4;

  const _Float16* Qbase = Qh + ((size_t)h * T_SEQ + qt * 16) * DH;
  v16h aq0 = load_a_frag(Qbase, DH, 0, 0);
  v16h aq1 = load_a_frag(Qbase, DH, 0, 32);

  const _Float16* Kbase = Kh + (size_t)(h & 15) * T_SEQ * DH;
  const float*    dbase = denom + (size_t)(h & 15) * T_SEQ;

  v8f zero = {};
  v8f cacc[4] = {zero, zero, zero, zero};
  v8f rowsum = {};
  __shared__ __align__(32) _Float16 wlds[16 * 32];

  int nchunks = (qt + 2) >> 1;           // ceil((16*qt+16)/32)
  for (int ch = 0; ch < nchunks; ++ch) {
    int s0c = ch * 32;
#pragma unroll
    for (int tt = 0; tt < 2; ++tt) {
      int st = ch * 2 + tt;
      int s0 = s0c + tt * 16;
      if (st <= qt) {                     // uniform branch (scalar)
        const _Float16* kp = Kbase + (size_t)(s0 + n) * DH + half * 16;
        v16h bk0 = *(const v16h*)kp;         // d = half*16 + j
        v16h bk1 = *(const v16h*)(kp + 32);  // d = 32 + half*16 + j
        v8f sc = wmma_f16(aq0, bk0, zero);
        sc = wmma_f16(aq1, bk1, sc);
        float invdn = ATTN_SCALE / dbase[s0 + n];
#pragma unroll
        for (int r = 0; r < 8; ++r) {
          int t = qt * 16 + r + half * 8;
          int s = s0 + n;
          float w = 0.f;
          if (s <= t) {
            float z = sc[r] * invdn;
            float sp = (z > 20.f) ? z : log1pf(__expf(z));
            w = sp / (1.f + __expf(-GATE * sp));   // sp * sigmoid(GATE*sp)
            rowsum[r] += w;
          }
          wlds[(r + half * 8) * 32 + tt * 16 + n] = (_Float16)w;
        }
      } else {
#pragma unroll
        for (int r = 0; r < 8; ++r)
          wlds[(r + half * 8) * 32 + tt * 16 + n] = (_Float16)0.f;
      }
    }
    __syncthreads();
    v16h aw = load_a_frag(wlds, 32, 0, 0);   // w as 16x32 A fragment
    __syncthreads();
#pragma unroll
    for (int dt = 0; dt < 4; ++dt) {
      // b[j] = V[s0c+kh*16+j][h*64+dt*16+n] = Vt[h*64+dt*16+n][s0c+kh*16+j]
      const _Float16* vp = Vt + (size_t)(h * DH + dt * 16 + n) * T_SEQ
                              + s0c + half * 16;
      v16h bv = *(const v16h*)vp;
      cacc[dt] = wmma_f16(aw, bv, cacc[dt]);
    }
  }

  // Row-sum reduction across the 16 lanes of each half-group, then finalize.
  float sink = tanhf(sink_scalars[h]) + 1e-6f;
  float alpha[8];
#pragma unroll
  for (int r = 0; r < 8; ++r) {
    float v = rowsum[r];
    v += __shfl_xor(v, 1);
    v += __shfl_xor(v, 2);
    v += __shfl_xor(v, 4);
    v += __shfl_xor(v, 8);
    alpha[r] = 1.f / (v + sink + 1e-6f);
  }
  int colbase = (h >> 4) * DMODEL + (h & 15) * DH;  // (branch, head) -> column
#pragma unroll
  for (int dt = 0; dt < 4; ++dt) {
    int d = dt * 16 + n;
    float vnl = v_nulls[(size_t)h * DH + d] * sink;
#pragma unroll
    for (int r = 0; r < 8; ++r) {
      int t = qt * 16 + r + half * 8;
      float val = alpha[r] * (cacc[dt][r] + vnl);
      ctxh[(size_t)t * HQDIM + colbase + d] = (_Float16)val;
    }
  }
}

// ---------------------------------------------------------------------------
// Host launch
// ---------------------------------------------------------------------------
extern "C" void kernel_launch(void* const* d_in, const int* in_sizes, int n_in,
                              void* d_out, int out_size, void* d_ws, size_t ws_size,
                              hipStream_t stream) {
  const float* X    = (const float*)d_in[0];
  const float* Wq   = (const float*)d_in[1];
  const float* bq   = (const float*)d_in[2];
  const float* Wk   = (const float*)d_in[3];
  const float* bk   = (const float*)d_in[4];
  const float* Wv   = (const float*)d_in[5];
  const float* bv   = (const float*)d_in[6];
  const float* sink = (const float*)d_in[7];
  const float* vnul = (const float*)d_in[8];
  const float* Wo   = (const float*)d_in[9];
  const float* bo   = (const float*)d_in[10];

  char* ws = (char*)d_ws;
  size_t off = 0;
  auto alloc = [&](size_t bytes) {
    char* p = ws + off;
    off += (bytes + 255) & ~(size_t)255;
    return p;
  };

  _Float16* Xh   = (_Float16*)alloc((size_t)T_SEQ * DMODEL * 2);
  _Float16* Wqt  = (_Float16*)alloc((size_t)DMODEL * HQDIM * 2);   // (4096,1024)
  _Float16* Wkt  = (_Float16*)alloc((size_t)DMODEL * DMODEL * 2);  // (1024,1024)
  _Float16* Wvt  = (_Float16*)alloc((size_t)DMODEL * HQDIM * 2);   // (4096,1024)
  _Float16* Wot  = (_Float16*)alloc((size_t)DMODEL * DMODEL * 2);  // (1024,1024)
  float*    Qf   = (float*)   alloc((size_t)T_SEQ * HQDIM * 4);
  float*    Kf   = (float*)   alloc((size_t)T_SEQ * DMODEL * 4);
  _Float16* Qh   = (_Float16*)alloc((size_t)HTOT * T_SEQ * DH * 2);
  _Float16* Kh   = (_Float16*)alloc((size_t)NHEAD * T_SEQ * DH * 2);
  _Float16* Vt   = (_Float16*)alloc((size_t)HQDIM * T_SEQ * 2);    // (4096,1024)
  float*    dnm  = (float*)   alloc((size_t)NHEAD * T_SEQ * 4);
  _Float16* Ctx  = (_Float16*)alloc((size_t)T_SEQ * HQDIM * 2);
  (void)ws_size; (void)in_sizes; (void)n_in; (void)out_size;

  const int CT = 256;
  dim3 tb(32, 8);
  cvt_f16<<<dim3((1048576 + CT - 1) / CT), CT, 0, stream>>>(X, Xh, 1048576);
  cvt_f16_t<<<dim3(HQDIM / 32, DMODEL / 32), tb, 0, stream>>>(Wq, DMODEL, HQDIM, Wqt);
  cvt_f16_t<<<dim3(DMODEL / 32, DMODEL / 32), tb, 0, stream>>>(Wk, DMODEL, DMODEL, Wkt);
  cvt_f16_t<<<dim3(HQDIM / 32, DMODEL / 32), tb, 0, stream>>>(Wv, DMODEL, HQDIM, Wvt);
  cvt_f16_t<<<dim3(DMODEL / 32, DMODEL / 32), tb, 0, stream>>>(Wo, DMODEL, DMODEL, Wot);

  const int NOMASK = 0x7FFFFFFF;
  // Q = X @ Wq + bq  -> f32 (1024 x 4096)
  gemm_wmma<<<dim3(HQDIM / 64, T_SEQ / 16), 32, 0, stream>>>(
      Xh, Wqt, bq, Qf, nullptr, DMODEL, DMODEL, DMODEL, HQDIM, NOMASK, 1.f, 0);
  // K = X @ Wk + bk  -> f32 (1024 x 1024)
  gemm_wmma<<<dim3(DMODEL / 64, T_SEQ / 16), 32, 0, stream>>>(
      Xh, Wkt, bk, Kf, nullptr, DMODEL, DMODEL, DMODEL, DMODEL, NOMASK, 1.f, 0);
  // V = X @ Wv + bv  -> f16 transposed (4096 x 1024)
  gemm_wmma<<<dim3(HQDIM / 64, T_SEQ / 16), 32, 0, stream>>>(
      Xh, Wvt, bv, nullptr, Vt, DMODEL, DMODEL, DMODEL, T_SEQ, NOMASK, 1.f, 1);

  // RoPE to per-head f16 layouts; K path also produces denom
  rope_kernel<<<dim3(T_SEQ, HTOT),  32, 0, stream>>>(Qf, HQDIM,  Qh, nullptr);
  rope_kernel<<<dim3(T_SEQ, NHEAD), 32, 0, stream>>>(Kf, DMODEL, Kh, dnm);

  // Streaming attention -> context f16 (T, 4096) laid out (branch, head, d)
  attn_kernel<<<dim3(T_SEQ / 16, HTOT), 32, 0, stream>>>(
      Qh, Kh, Vt, dnm, sink, vnul, Ctx);

  // y = mean_br(ctx_br @ Wo) + bo  == 0.25 * (Ctx(T,4096) @ tiled Wo) + bo
  gemm_wmma<<<dim3(DMODEL / 64, T_SEQ / 16), 32, 0, stream>>>(
      Ctx, Wot, bo, (float*)d_out, nullptr, HQDIM, HQDIM, DMODEL, DMODEL,
      1023, 0.25f, 0);
}
